// SessionGraph_26139170964316
// MI455X (gfx1250) — matrix operand order repaired and checked by reading, not compiled, not run on hardware
//
#include <hip/hip_runtime.h>
#include <hip/hip_bf16.h>
#include <math.h>

#define N_NODE 100000
#define H 128
#define NI 4
#define B 4096
#define T 50
#define KDIM 256            // fused K: [x(128) | h(128)]
#define NDIM 512            // fused N: [r(128) | i(128) | i_n(128) | h_n(128)]
#define NT (NDIM / 16)      // 32 n-tiles
#define KT (KDIM / 32)      // 8 k-tiles

typedef __attribute__((ext_vector_type(16))) __bf16 v16bf;
typedef __attribute__((ext_vector_type(8)))  float  v8f;

__device__ __forceinline__ unsigned short f2bf(float f) {
  union { float f; unsigned u; } x; x.f = f;
  unsigned r = x.u + 0x7FFFu + ((x.u >> 16) & 1u);   // round-to-nearest-even
  return (unsigned short)(r >> 16);
}
__device__ __forceinline__ float sigm(float x) { return 1.0f / (1.0f + __expf(-x)); }

// ---------------------------------------------------------------------------
// Repack weights into bf16 WMMA B-fragment layout + fused biases.
// Wc[i][kt][nt][lane][e]: lane half (lane>>4) selects K 0-15 vs 16-31; elem e = K
// within half. n = nt*16 + (lane&15). Columns: [0,256)=r,i fused over K=256;
// [256,384)=i_n (x half only); [384,512)=h_n (h half only).
// ---------------------------------------------------------------------------
__global__ void prep_kernel(const float* __restrict__ w_ih, const float* __restrict__ w_hh,
                            const float* __restrict__ b_ih, const float* __restrict__ b_hh,
                            unsigned short* __restrict__ Wc, float* __restrict__ bsum) {
  const int total = NI * KT * NT * 32 * 16;   // 524288 fragment elements
  for (int idx = blockIdx.x * blockDim.x + threadIdx.x;
       idx < total + NI * NDIM; idx += gridDim.x * blockDim.x) {
    if (idx < total) {
      int e    = idx & 15;
      int lane = (idx >> 4) & 31;
      int nt   = (idx >> 9) & 31;
      int kt   = (idx >> 14) & 7;
      int i    = idx >> 17;
      int klocal = ((lane >> 4) * 16) + e;      // B layout: half selects K group
      int k = kt * 32 + klocal;                 // 0..255
      int n = nt * 16 + (lane & 15);            // 0..511
      float v = 0.0f;
      if (n < 256) {            // r and i gates: fused x+h contribution
        v = (k < 128) ? w_ih[(i * 384 + n) * H + k]
                      : w_hh[(i * 384 + n) * H + (k - 128)];
      } else if (n < 384) {     // i_n: x contribution only
        if (k < 128) v = w_ih[(i * 384 + n) * H + k];
      } else {                  // h_n: h contribution only (rows 256..383)
        if (k >= 128) v = w_hh[(i * 384 + (n - 128)) * H + (k - 128)];
      }
      Wc[idx] = f2bf(v);
    } else {
      int j = idx - total;                      // 0..2047
      int i = j >> 9;
      int n = j & (NDIM - 1);
      float v;
      if (n < 256)      v = b_ih[i * 384 + n] + b_hh[i * 384 + n];
      else if (n < 384) v = b_ih[i * 384 + n];
      else              v = b_hh[i * 384 + (n - 128)];
      bsum[j] = v;
    }
  }
}

// ---------------------------------------------------------------------------
// Gather embedding rows -> bf16 x[t][b][:], and interest softmax -> g_t[i][t][b].
// One wave32 per (t,b) row; 8 rows per 256-thread block.
// ---------------------------------------------------------------------------
__global__ __launch_bounds__(256) void gather_kernel(
    const int* __restrict__ inputs, const float* __restrict__ emb,
    const float* __restrict__ w_int, unsigned short* __restrict__ xbf,
    float* __restrict__ gt) {
  int wave = threadIdx.x >> 5, lane = threadIdx.x & 31;
  long r = (long)blockIdx.x * 8 + wave;     // row over T*B
  int t = (int)(r / B), b = (int)(r % B);
  int node = inputs[b * T + t];
  const float* e = emb + (long)node * H;
  float e0 = e[lane], e1 = e[32 + lane], e2 = e[64 + lane], e3 = e[96 + lane];
  unsigned short* xr = xbf + ((long)t * B + b) * H;
  xr[lane] = f2bf(e0); xr[32 + lane] = f2bf(e1);
  xr[64 + lane] = f2bf(e2); xr[96 + lane] = f2bf(e3);
  float s[4];
  #pragma unroll
  for (int j = 0; j < 4; ++j) {
    const float* w = w_int + j * H;
    float a = e0 * w[lane] + e1 * w[32 + lane] + e2 * w[64 + lane] + e3 * w[96 + lane];
    #pragma unroll
    for (int m = 16; m >= 1; m >>= 1) a += __shfl_xor(a, m, 32);
    s[j] = a;
  }
  if (lane == 0) {
    float mx = fmaxf(fmaxf(s[0], s[1]), fmaxf(s[2], s[3]));
    float p[4], sum = 0.0f;
    #pragma unroll
    for (int j = 0; j < 4; ++j) { p[j] = __expf((s[j] - mx) * 10.0f); sum += p[j]; }
    float inv = 1.0f / sum;
    #pragma unroll
    for (int j = 0; j < 4; ++j) gt[((long)j * T + t) * B + b] = p[j] * inv;
  }
}

// ---------------------------------------------------------------------------
// One recurrence step for one interest, one 16-row batch tile.
// 8 waves: each owns 4 N-tiles of the fused 512-wide gate GEMM, K=256 (8 WMMAs
// of v_wmma_f32_16x16x32_bf16 per N-tile). Then gate math + blend.
// ---------------------------------------------------------------------------
__global__ __launch_bounds__(256) void gru_step_kernel(
    const unsigned short* __restrict__ Wc, const float* __restrict__ bsum,
    const unsigned short* __restrict__ xbf, const float* __restrict__ gt,
    const float* __restrict__ hin, float* __restrict__ hout, int t) {
  __shared__ unsigned short Ash[16][KDIM];  // 16 x 256 bf16 = 8 KB
  __shared__ float Gsh[16][NDIM];           // 16 x 512 f32  = 32 KB
  const int i  = blockIdx.y;
  const int b0 = blockIdx.x * 16;
  const int tid = threadIdx.x;

  const unsigned short* xrow = xbf + ((long)t * B + b0) * H;
  const float* hrow = hin + ((long)i * B + b0) * H;
  for (int idx = tid; idx < 16 * H; idx += 256) {
    int r = idx >> 7, c = idx & 127;
    Ash[r][c]       = xrow[r * H + c];            // x already bf16
    Ash[r][128 + c] = f2bf(hrow[r * H + c]);      // h: f32 -> bf16
  }
  __syncthreads();

  const int wave = tid >> 5, lane = tid & 31;
  const int half = lane >> 4;         // lane group selects K sub-block
  const int arow = lane & 15;         // A fragment row

  v8f acc[4];
  #pragma unroll
  for (int q = 0; q < 4; ++q) {
    int n = (wave * 4 + q) * 16 + (lane & 15);
    float bv = bsum[i * NDIM + n];
    #pragma unroll
    for (int v = 0; v < 8; ++v) acc[q][v] = bv;   // C initialized with bias
  }

  for (int kt = 0; kt < KT; ++kt) {
    union { v16bf v; unsigned short u[16]; } af;
    int kb = kt * 32;
    // A in this code's contiguous k order: lanes 0-15 hold K {0..15}, lanes 16-31 {16..31} (matches Wc B packing)
    #pragma unroll
    for (int e2 = 0; e2 < 8; ++e2) af.u[e2]     = Ash[arow][kb + half * 16 + e2];
    #pragma unroll
    for (int e2 = 0; e2 < 8; ++e2) af.u[8 + e2] = Ash[arow][kb + half * 16 + 8 + e2];
    #pragma unroll
    for (int q = 0; q < 4; ++q) {
      int nt = wave * 4 + q;
      const unsigned short* bp = Wc + ((((long)i * KT + kt) * NT + nt) * 32 + lane) * 16;
      union { v16bf v; unsigned short u[16]; } bfr;
      *(uint4*)&bfr.u[0] = *(const uint4*)(bp);       // 32B contiguous B fragment
      *(uint4*)&bfr.u[8] = *(const uint4*)(bp + 8);
      acc[q] = __builtin_amdgcn_wmma_f32_16x16x32_bf16(
          false, af.v, false, bfr.v, (short)0, acc[q], false, false);
    }
  }

  // C/D layout: VGPR v, lane L -> row v + (L>=16)*8, col (L&15)
  #pragma unroll
  for (int q = 0; q < 4; ++q) {
    int n  = (wave * 4 + q) * 16 + (lane & 15);
    int rb = half * 8;
    #pragma unroll
    for (int v = 0; v < 8; ++v) Gsh[rb + v][n] = acc[q][v];
  }
  __syncthreads();

  const float* gvt = gt + ((long)i * T + t) * B + b0;
  for (int idx = tid; idx < 16 * H; idx += 256) {
    int r = idx >> 7, c = idx & 127;
    float rg = sigm(Gsh[r][c]);
    float ig = sigm(Gsh[r][128 + c]);
    float ng = tanhf(Gsh[r][256 + c] + rg * Gsh[r][384 + c]);
    float gv = gvt[r];
    float cg = (gv > 0.01f ? gv : 0.0f) * ig;
    float hp = hrow[r * H + c];
    hout[((long)i * B + b0 + r) * H + c] = (1.0f - cg) * hp + cg * ng;
  }
}

// [4,B,H] -> [B,4,H]
__global__ void finalize_kernel(const float* __restrict__ hfin, float* __restrict__ out) {
  for (long idx = (long)blockIdx.x * blockDim.x + threadIdx.x;
       idx < (long)NI * B * H; idx += (long)gridDim.x * blockDim.x) {
    long c = idx & (H - 1);
    long i = (idx >> 7) & (NI - 1);
    long b = idx >> 9;
    out[idx] = hfin[(i * B + b) * H + c];
  }
}

extern "C" void kernel_launch(void* const* d_in, const int* in_sizes, int n_in,
                              void* d_out, int out_size, void* d_ws, size_t ws_size,
                              hipStream_t stream) {
  const int*   inputs = (const int*)d_in[0];
  const float* emb    = (const float*)d_in[1];
  const float* w_int  = (const float*)d_in[2];
  const float* w_ih   = (const float*)d_in[3];
  const float* w_hh   = (const float*)d_in[4];
  const float* b_ih   = (const float*)d_in[5];
  const float* b_hh   = (const float*)d_in[6];
  const float* h0     = (const float*)d_in[7];
  (void)in_sizes; (void)n_in; (void)out_size; (void)ws_size;

  char* ws = (char*)d_ws;
  size_t off = 0;
  auto wsalloc = [&](size_t bytes) -> void* {
    void* p = ws + off;
    off = (off + bytes + 255) & ~(size_t)255;
    return p;
  };
  unsigned short* Wc  = (unsigned short*)wsalloc((size_t)NI * KT * NT * 32 * 16 * 2); // 1 MB
  float* bsum         = (float*)wsalloc((size_t)NI * NDIM * 4);                       // 8 KB
  unsigned short* xbf = (unsigned short*)wsalloc((size_t)T * B * H * 2);              // 52 MB
  float* gtb          = (float*)wsalloc((size_t)NI * T * B * 4);                      // 3.3 MB
  float* hA           = (float*)wsalloc((size_t)NI * B * H * 4);                      // 8.4 MB
  float* hB           = (float*)wsalloc((size_t)NI * B * H * 4);                      // 8.4 MB

  prep_kernel<<<512, 256, 0, stream>>>(w_ih, w_hh, b_ih, b_hh, Wc, bsum);
  hipMemcpyAsync(hA, h0, (size_t)NI * B * H * 4, hipMemcpyDeviceToDevice, stream);
  gather_kernel<<<(T * B) / 8, 256, 0, stream>>>(inputs, emb, w_int, xbf, gtb);

  for (int t = 0; t < T; ++t) {
    const float* hin = (t & 1) ? hB : hA;
    float* hout      = (t & 1) ? hA : hB;
    gru_step_kernel<<<dim3(B / 16, NI), 256, 0, stream>>>(Wc, bsum, xbf, gtb, hin, hout, t);
  }
  // T=50 is even -> final state lands in hA
  finalize_kernel<<<1024, 256, 0, stream>>>(hA, (float*)d_out);
}
